// AttentionBlock_80960133530085
// MI455X (gfx1250) — matrix operand order repaired and checked by reading, not compiled, not run on hardware
//
#include <hip/hip_runtime.h>
#include <hip/hip_bf16.h>

typedef __attribute__((ext_vector_type(16))) __bf16       v16bf;
typedef __attribute__((ext_vector_type(8)))  float        v8f;
typedef __attribute__((ext_vector_type(4)))  unsigned int v4u;

union BFrag {
    v16bf          v;
    __bf16         h[16];
    unsigned short u[16];
    unsigned int   w[8];
};

// hardware f32 -> bf16 (RNE) via native cast; returned as raw bits
__device__ inline unsigned short f32_to_bf16u(float f) {
    union { __bf16 b; unsigned short s; } cv;
    cv.b = (__bf16)f;
    return cv.s;
}

__device__ inline v8f wmma_bf16(v16bf a, v16bf b, v8f c) {
    return __builtin_amdgcn_wmma_f32_16x16x32_bf16(
        /*neg_a=*/false, a, /*neg_b=*/false, b,
        /*c_mod=*/(short)0, c, /*reuse_a=*/false, /*reuse_b=*/false);
}

// Load a 16-element-per-lane bf16 fragment as two 16-byte chunks at bf16
// offsets off0 / off1 from p (p = row start incl. kbase for this lane).
// A layout: off0 = half*16, off1 = half*16 + 8
// B layout: off0 = half*16, off1 = half*16 + 8
__device__ inline v16bf ldfrag16(const unsigned short* p, int off0, int off1) {
    BFrag f;
    v4u a = *(const v4u*)(p + off0);
    v4u b = *(const v4u*)(p + off1);
    f.w[0] = a.x; f.w[1] = a.y; f.w[2] = a.z; f.w[3] = a.w;
    f.w[4] = b.x; f.w[5] = b.y; f.w[6] = b.z; f.w[7] = b.w;
    return f.v;
}

// A-fragment from contiguous f32 row (W): native convert to bf16.
__device__ inline v16bf ldfragA_f32(const float* p, int half) {
    BFrag f;
    const float* q0 = p + half * 16;
    const float* q1 = p + half * 16 + 8;
#pragma unroll
    for (int i = 0; i < 8; ++i) {
        f.h[i]     = (__bf16)q0[i];
        f.h[8 + i] = (__bf16)q1[i];
    }
    return f.v;
}

// B-fragment from f32 with column-stride access (x): element k at
// p[(kbase + half*16 + i)*stride + col]. Coalesced across the 16 lanes.
__device__ inline v16bf ldfragB_f32_strided(const float* p, int stride, int col,
                                            int kbase, int half) {
    BFrag f;
#pragma unroll
    for (int i = 0; i < 16; ++i)
        f.h[i] = (__bf16)p[(size_t)(kbase + half * 16 + i) * stride + col];
    return f.v;
}

// ---------------------------------------------------------------------------
// Kernel 1: qkv = W @ x  (per batch), written as bf16 into workspace:
//   qT [head][n][64] * 0.125, kT [head][n][64] * 0.125, v [head][64][n]
// ---------------------------------------------------------------------------
__global__ __launch_bounds__(256) void qkv_gemm_kernel(
    const float* __restrict__ x, const float* __restrict__ W,
    unsigned short* __restrict__ qT, unsigned short* __restrict__ kT,
    unsigned short* __restrict__ vv) {
    const int C = 512, N = 2048;
    int wave = threadIdx.x >> 5, lane = threadIdx.x & 31;
    int half = lane >> 4, ln = lane & 15;
    int n_tile = blockIdx.x;              // 0..127
    int o_tile = blockIdx.y * 8 + wave;   // 0..95
    int b = blockIdx.z;                   // 0..7

    const float* xb   = x + (size_t)b * C * N;
    const float* wrow = W + (size_t)(o_tile * 16 + ln) * C;  // A row for lane
    int ncol = n_tile * 16 + ln;                             // B col for lane

    v8f acc = {};
#pragma unroll 2
    for (int k0 = 0; k0 < C; k0 += 32) {
        v16bf a  = ldfragA_f32(wrow + k0, half);
        v16bf bm = ldfragB_f32_strided(xb, N, ncol, k0, half);
        acc = wmma_bf16(a, bm, acc);
    }

    // whole 16-row tile lies in a single 64-aligned q/k/v section;
    // these are wave-uniform -> force scalar to get uniform branches.
    int base_o = o_tile * 16;
    int hh    = __builtin_amdgcn_readfirstlane(base_o / 192);
    int rsec  = __builtin_amdgcn_readfirstlane(base_o % 192);
    int sec   = rsec / 64;       // 0=q 1=k 2=v
    int cbase = rsec % 64;
    int head  = b * 8 + hh;
    int n = n_tile * 16 + ln;

    if (sec == 0) {
#pragma unroll
        for (int r = 0; r < 8; ++r) {
            int c = cbase + r + 8 * half;
            qT[((size_t)head * N + n) * 64 + c] = f32_to_bf16u(acc[r] * 0.125f);
        }
    } else if (sec == 1) {
#pragma unroll
        for (int r = 0; r < 8; ++r) {
            int c = cbase + r + 8 * half;
            kT[((size_t)head * N + n) * 64 + c] = f32_to_bf16u(acc[r] * 0.125f);
        }
    } else {
#pragma unroll
        for (int r = 0; r < 8; ++r) {
            int c = cbase + r + 8 * half;
            vv[((size_t)head * 64 + c) * N + n] = f32_to_bf16u(acc[r]);
        }
    }
}

// ---------------------------------------------------------------------------
// Kernel 2: flash attention. Block = 8 waves; wave owns 16 query rows.
// s-loop in tiles of 32: 4 WMMAs for S, online softmax, P via LDS, 4 WMMAs
// for P*V^T into four 16x16 f32 accumulators.
// ---------------------------------------------------------------------------
__global__ __launch_bounds__(256) void flash_attn_kernel(
    const unsigned short* __restrict__ qT, const unsigned short* __restrict__ kT,
    const unsigned short* __restrict__ vv, float* __restrict__ out) {
    const int N = 2048, D = 64;
    __shared__ unsigned short Pbuf[8][16 * 32];

    int wave = threadIdx.x >> 5, lane = threadIdx.x & 31;
    int half = lane >> 4, ln = lane & 15;
    int h  = blockIdx.y;
    int t0 = blockIdx.x * 128 + wave * 16;

    const unsigned short* q = qT + (size_t)h * N * D;
    const unsigned short* k = kT + (size_t)h * N * D;
    const unsigned short* v = vv + (size_t)h * D * N;
    unsigned short* P = &Pbuf[wave][0];

    // Q fragments (A layout), row = t0 + ln, K(c) = 0..31 and 32..63
    const unsigned short* qrow = q + (size_t)(t0 + ln) * D;
    v16bf qf0 = ldfrag16(qrow + 0,  half * 16, half * 16 + 8);
    v16bf qf1 = ldfrag16(qrow + 32, half * 16, half * 16 + 8);

    float mrow[8], lrow[8];
    v8f o0 = {}, o1 = {}, o2 = {}, o3 = {};
#pragma unroll
    for (int r = 0; r < 8; ++r) { mrow[r] = -3.0e38f; lrow[r] = 0.f; }

    for (int s0 = 0; s0 < N; s0 += 32) {
        // prefetch next s-tile of K and V into cache (global_prefetch_b8)
        if (s0 + 32 < N) {
            __builtin_prefetch(k + (size_t)(s0 + 32 + ln) * D, 0, 3);
            __builtin_prefetch(k + (size_t)(s0 + 48 + ln) * D, 0, 3);
            __builtin_prefetch(v + (size_t)(ln +  0) * N + s0 + 32, 0, 3);
            __builtin_prefetch(v + (size_t)(ln + 16) * N + s0 + 32, 0, 3);
            __builtin_prefetch(v + (size_t)(ln + 32) * N + s0 + 32, 0, 3);
            __builtin_prefetch(v + (size_t)(ln + 48) * N + s0 + 32, 0, 3);
        }

        // K fragments (B layout): column s = s0+ln and s0+16+ln
        const unsigned short* kr0 = k + (size_t)(s0 + ln) * D;
        const unsigned short* kr1 = k + (size_t)(s0 + 16 + ln) * D;
        v16bf kf00 = ldfrag16(kr0 + 0,  half * 16, half * 16 + 8);
        v16bf kf01 = ldfrag16(kr0 + 32, half * 16, half * 16 + 8);
        v16bf kf10 = ldfrag16(kr1 + 0,  half * 16, half * 16 + 8);
        v16bf kf11 = ldfrag16(kr1 + 32, half * 16, half * 16 + 8);

        v8f sa = {}, sb = {};
        sa = wmma_bf16(qf0, kf00, sa);
        sa = wmma_bf16(qf1, kf01, sa);
        sb = wmma_bf16(qf0, kf10, sb);
        sb = wmma_bf16(qf1, kf11, sb);

        // online softmax over the 32 new columns; row m = r + 8*half,
        // row stats reduced across the 16 lanes of this half.
#pragma unroll
        for (int r = 0; r < 8; ++r) {
            float mx = fmaxf(sa[r], sb[r]);
            mx = fmaxf(mx, __shfl_xor(mx, 1, 32));
            mx = fmaxf(mx, __shfl_xor(mx, 2, 32));
            mx = fmaxf(mx, __shfl_xor(mx, 4, 32));
            mx = fmaxf(mx, __shfl_xor(mx, 8, 32));
            float nm = fmaxf(mrow[r], mx);
            float p0 = __expf(sa[r] - nm);
            float p1 = __expf(sb[r] - nm);
            P[(r + 8 * half) * 32 + ln]      = f32_to_bf16u(p0);
            P[(r + 8 * half) * 32 + 16 + ln] = f32_to_bf16u(p1);
            float ps = p0 + p1;
            ps += __shfl_xor(ps, 1, 32);
            ps += __shfl_xor(ps, 2, 32);
            ps += __shfl_xor(ps, 4, 32);
            ps += __shfl_xor(ps, 8, 32);
            float corr = __expf(mrow[r] - nm);
            lrow[r] = lrow[r] * corr + ps;
            mrow[r] = nm;
            o0[r] *= corr; o1[r] *= corr; o2[r] *= corr; o3[r] *= corr;
        }

        // same-wave LDS ordering: make P stores visible before A-layout reads
        asm volatile("s_wait_dscnt 0" ::: "memory");
        const unsigned short* prow = P + ln * 32;           // row t_local = ln
        v16bf pf = ldfrag16(prow, half * 16, half * 16 + 8); // A layout, K(s)=0..31

        // V fragments (B layout): B[k=s][n=c], contiguous along s
        const unsigned short* v0 = v + (size_t)(ln +  0) * N + s0;
        const unsigned short* v1 = v + (size_t)(ln + 16) * N + s0;
        const unsigned short* v2 = v + (size_t)(ln + 32) * N + s0;
        const unsigned short* v3 = v + (size_t)(ln + 48) * N + s0;
        o0 = wmma_bf16(pf, ldfrag16(v0, half * 16, half * 16 + 8), o0);
        o1 = wmma_bf16(pf, ldfrag16(v1, half * 16, half * 16 + 8), o1);
        o2 = wmma_bf16(pf, ldfrag16(v2, half * 16, half * 16 + 8), o2);
        o3 = wmma_bf16(pf, ldfrag16(v3, half * 16, half * 16 + 8), o3);
    }

    // epilogue: out[b][hh*64 + c][t] = O[t][c] / l[t]
    int b = h >> 3, hh = h & 7;
    float* outp = out + ((size_t)b * 512 + (size_t)hh * 64) * N;
#pragma unroll
    for (int r = 0; r < 8; ++r) {
        float inv = 1.0f / lrow[r];
        size_t t = (size_t)(t0 + r + 8 * half);
        outp[(size_t)(ln +  0) * N + t] = o0[r] * inv;
        outp[(size_t)(ln + 16) * N + t] = o1[r] * inv;
        outp[(size_t)(ln + 32) * N + t] = o2[r] * inv;
        outp[(size_t)(ln + 48) * N + t] = o3[r] * inv;
    }
}

extern "C" void kernel_launch(void* const* d_in, const int* in_sizes, int n_in,
                              void* d_out, int out_size, void* d_ws, size_t ws_size,
                              hipStream_t stream) {
    const float* x = (const float*)d_in[0];   // [8,512,2048] f32
    const float* W = (const float*)d_in[1];   // [1536,512]  f32
    float* out = (float*)d_out;               // [8,512,2048] f32

    // workspace: qT(16MB) | kT(16MB) | v(16MB), all bf16
    unsigned short* qT = (unsigned short*)d_ws;
    unsigned short* kT = qT + (size_t)64 * 2048 * 64;
    unsigned short* vv = kT + (size_t)64 * 2048 * 64;

    dim3 g1(128, 12, 8);                      // n_tiles, o_tile_groups, batch
    qkv_gemm_kernel<<<g1, 256, 0, stream>>>(x, W, qT, kT, vv);

    dim3 g2(16, 64);                          // t tiles of 128, heads
    flash_attn_kernel<<<g2, 256, 0, stream>>>(qT, kT, vv, out);
}
